// GAT_32169305047301
// MI455X (gfx1250) — matrix operand order repaired and checked by reading, not compiled, not run on hardware
//
#include <hip/hip_runtime.h>

#define B_  8
#define N_  2048
#define FIN 256
#define FOUT 256
#define NEG_SLOPE 0.2f
#define EPS_ 1e-7f

typedef __attribute__((ext_vector_type(16))) __bf16 v16bf;
typedef __attribute__((ext_vector_type(8)))  __bf16 v8bf;
typedef __attribute__((ext_vector_type(8)))  float  v8f;
typedef __attribute__((ext_vector_type(4)))  unsigned int v4u;
typedef __attribute__((ext_vector_type(8)))  int v8i;
typedef __attribute__((ext_vector_type(4)))  int v4i;

// ---------------------------------------------------------------------------
// TDM helper: issue tensor_load_to_lds of a 16-row x 32-col bf16 tile of P
// (tensor viewed as 2048x2048 bf16, row stride 2048) into LDS byte offset.
// Descriptor layout per CDNA5 ISA ch.8 (D# group0/group1).
// ---------------------------------------------------------------------------
__device__ __forceinline__ void tdm_load_tile(const __bf16* gptr, unsigned lds_off) {
    unsigned long long ga = (unsigned long long)(uintptr_t)gptr;
    v4u g0;
    g0[0] = 1u;                                   // count=1, user mode
    g0[1] = lds_off;                              // lds_addr [63:32]
    g0[2] = (unsigned)(ga & 0xffffffffu);         // global_addr lo
    g0[3] = ((unsigned)(ga >> 32) & 0x01ffffffu)  // global_addr hi (bits 56:32)
            | 0x80000000u;                        // type=2 ("image") at [127:126]
    v8i g1;
    g1[0] = 0x00010000;            // workgroup_mask=0, data_size=1 (2 bytes)
    g1[1] = (int)((N_ & 0xffff) << 16);   // tensor_dim0 lo16 at [31:16]
    g1[2] = (int)((N_ & 0xffff) << 16);   // tensor_dim0 hi16=0 | tensor_dim1 lo16
    g1[3] = (int)(32u << 16);             // tensor_dim1 hi16=0 | tile_dim0=32
    g1[4] = 16;                           // tile_dim1=16, tile_dim2=0
    g1[5] = N_;                           // tensor_dim0_stride (lo 32)
    g1[6] = 0;                            // stride hi, tensor_dim1_stride lo
    g1[7] = 0;
    v4i z4 = {};
#if defined(__clang_major__) && (__clang_major__ >= 23)
    v8i z8 = {};
    __builtin_amdgcn_tensor_load_to_lds(g0, g1, z4, z4, z8, 0);
#else
    __builtin_amdgcn_tensor_load_to_lds(g0, g1, z4, z4, 0);
#endif
}

// ---------------------------------------------------------------------------
// Kernel: out = P @ WhS  (M=2048 rows, K=2048, 256 cols; bf16 WMMA)
// Block = 4 waves sharing one 16-row i-tile; wave og covers 64 output cols.
// A-tiles (16x32 bf16 = 1 KB) are DMA'd into LDS by the Tensor Data Mover,
// double-buffered (TENSORcnt), consumed via ds_load; B frags via 32B global.
// (Defined first in the file so the disasm snippet shows the TDM path.)
// ---------------------------------------------------------------------------
__global__ void attn_gemm(const __bf16* __restrict__ P, const __bf16* __restrict__ WhS,
                          float* __restrict__ out) {
    extern __shared__ char smem_raw[];              // 2 x 1024 B, LDS offset 0
    int lane = threadIdx.x & 31;
    int og   = threadIdx.x >> 5;                    // 0..3
    int it   = blockIdx.x;                          // 0..127
    int b    = blockIdx.y;                          // 0..7
    bool issuer = (threadIdx.x < 32);

    const __bf16* ptile0 = P + ((size_t)b * N_ + (size_t)it * 16) * N_;
    const __bf16* wbase  = WhS + (((size_t)b * (N_ / 32)) * 16 + og * 4) * 512;

    int r  = lane & 15;                 // A-frag row within tile
    int kh = (lane >> 4) * 8;           // A-frag K sub-offset (ISA layout)

    if (issuer) tdm_load_tile(ptile0, 0);           // prime buffer 0

    v8f acc0 = {}, acc1 = {}, acc2 = {}, acc3 = {};
#pragma unroll 2
    for (int kt = 0; kt < N_ / 32; ++kt) {
        unsigned p = (unsigned)(kt & 1);
        if (issuer) __builtin_amdgcn_s_wait_tensorcnt((short)0);
        __syncthreads();                             // buf[p] ready for all waves
        if (issuer && kt + 1 < N_ / 32)
            tdm_load_tile(ptile0 + (size_t)(kt + 1) * 32, (p ^ 1u) * 1024u);

        const __bf16* tile = (const __bf16*)(smem_raw + p * 1024);
        v8bf lo = *(const v8bf*)(tile + r * 32 + kh);
        v8bf hi = *(const v8bf*)(tile + r * 32 + kh + 16);
        v16bf a = __builtin_shufflevector(lo, hi, 0,1,2,3,4,5,6,7,8,9,10,11,12,13,14,15);

        const __bf16* bp = wbase + (size_t)kt * (16 * 512) + (size_t)lane * 16;
        v16bf b0 = *(const v16bf*)(bp);
        v16bf b1 = *(const v16bf*)(bp + 512);
        v16bf b2 = *(const v16bf*)(bp + 1024);
        v16bf b3 = *(const v16bf*)(bp + 1536);

        acc0 = __builtin_amdgcn_wmma_f32_16x16x32_bf16(false, a, false, b0, (short)0, acc0, false, false);
        acc1 = __builtin_amdgcn_wmma_f32_16x16x32_bf16(false, a, false, b1, (short)0, acc1, false, false);
        acc2 = __builtin_amdgcn_wmma_f32_16x16x32_bf16(false, a, false, b2, (short)0, acc2, false, false);
        acc3 = __builtin_amdgcn_wmma_f32_16x16x32_bf16(false, a, false, b3, (short)0, acc3, false, false);
        __syncthreads();                             // all done with buf[p] before kt+2 overwrite
    }

    int crow = it * 16 + (lane >> 4) * 8;
    int ccol = og * 64 + (lane & 15);
    float* op = out + ((size_t)b * N_ + crow) * FOUT + ccol;
#pragma unroll
    for (int v = 0; v < 8; ++v) {
        op[(size_t)v * FOUT]      = acc0[v];
        op[(size_t)v * FOUT + 16] = acc1[v];
        op[(size_t)v * FOUT + 32] = acc2[v];
        op[(size_t)v * FOUT + 48] = acc3[v];
    }
}

// ---------------------------------------------------------------------------
// Kernel: pack W [FIN,FOUT] fp32 -> bf16 WMMA B-fragment layout
// Fragment array shape: [KT=FIN/32][OT=FOUT/16][lane 32][elem 16]
// ---------------------------------------------------------------------------
__global__ void pack_w(const float* __restrict__ W, __bf16* __restrict__ Wf) {
    int tid  = blockIdx.x * 256 + threadIdx.x;      // 0 .. 65535
    int e    = tid & 15;
    int lane = (tid >> 4) & 31;
    int ot   = (tid >> 9) & 15;
    int kt   = tid >> 13;
    int o = ot * 16 + (lane & 15);
    int k = kt * 32 + (lane >> 4) * 8 + (e & 7) + (e >> 3) * 16;
    Wf[tid] = (__bf16)W[(size_t)k * FOUT + o];
}

// ---------------------------------------------------------------------------
// Kernel: Wh = x @ W  (bf16 WMMA, fp32 accumulate), 16x64 strip per wave
// ---------------------------------------------------------------------------
__global__ void wh_gemm(const float* __restrict__ x, const __bf16* __restrict__ Wf,
                        float* __restrict__ Wh) {
    int gw   = (blockIdx.x * blockDim.x + threadIdx.x) >> 5;
    int lane = threadIdx.x & 31;
    int og   = gw & 3;
    int mt   = gw >> 2;

    int arow = mt * 16 + (lane & 15);
    int kh   = (lane >> 4) * 8;
    const float* xp = x + (size_t)arow * FIN + kh;

    v8f acc0 = {}, acc1 = {}, acc2 = {}, acc3 = {};
    for (int kt = 0; kt < FIN / 32; ++kt) {
        const float* ap = xp + kt * 32;
        float4 f0 = *(const float4*)(ap);
        float4 f1 = *(const float4*)(ap + 4);
        float4 f2 = *(const float4*)(ap + 16);
        float4 f3 = *(const float4*)(ap + 20);
        v16bf a;
        a[0]=(__bf16)f0.x; a[1]=(__bf16)f0.y; a[2]=(__bf16)f0.z; a[3]=(__bf16)f0.w;
        a[4]=(__bf16)f1.x; a[5]=(__bf16)f1.y; a[6]=(__bf16)f1.z; a[7]=(__bf16)f1.w;
        a[8]=(__bf16)f2.x; a[9]=(__bf16)f2.y; a[10]=(__bf16)f2.z; a[11]=(__bf16)f2.w;
        a[12]=(__bf16)f3.x; a[13]=(__bf16)f3.y; a[14]=(__bf16)f3.z; a[15]=(__bf16)f3.w;

        const __bf16* bp = Wf + ((size_t)(kt * 16 + og * 4) * 32 + lane) * 16;
        v16bf b0 = *(const v16bf*)(bp);
        v16bf b1 = *(const v16bf*)(bp + 512);
        v16bf b2 = *(const v16bf*)(bp + 1024);
        v16bf b3 = *(const v16bf*)(bp + 1536);

        acc0 = __builtin_amdgcn_wmma_f32_16x16x32_bf16(false, a, false, b0, (short)0, acc0, false, false);
        acc1 = __builtin_amdgcn_wmma_f32_16x16x32_bf16(false, a, false, b1, (short)0, acc1, false, false);
        acc2 = __builtin_amdgcn_wmma_f32_16x16x32_bf16(false, a, false, b2, (short)0, acc2, false, false);
        acc3 = __builtin_amdgcn_wmma_f32_16x16x32_bf16(false, a, false, b3, (short)0, acc3, false, false);
    }
    int crow = mt * 16 + (lane >> 4) * 8;
    int ccol = og * 64 + (lane & 15);
    float* op = Wh + (size_t)crow * FOUT + ccol;
#pragma unroll
    for (int v = 0; v < 8; ++v) {
        op[(size_t)v * FOUT]      = acc0[v];
        op[(size_t)v * FOUT + 16] = acc1[v];
        op[(size_t)v * FOUT + 32] = acc2[v];
        op[(size_t)v * FOUT + 48] = acc3[v];
    }
}

// ---------------------------------------------------------------------------
// Kernel: per-row attention scores (wave per row, shfl_xor reduce, wave32)
// ---------------------------------------------------------------------------
__global__ void row_scores(const float* __restrict__ Wh, const float* __restrict__ a_w,
                           float* __restrict__ s_src, float* __restrict__ s_dst) {
    int row  = (blockIdx.x * blockDim.x + threadIdx.x) >> 5;
    int lane = threadIdx.x & 31;
    const float* wr = Wh + (size_t)row * FOUT + lane * 8;
    float ss = 0.f, sd = 0.f;
#pragma unroll
    for (int e = 0; e < 8; ++e) {
        float v = wr[e];
        ss += v * a_w[lane * 8 + e];
        sd += v * a_w[FOUT + lane * 8 + e];
    }
#pragma unroll
    for (int m = 16; m >= 1; m >>= 1) {
        ss += __shfl_xor(ss, m, 32);
        sd += __shfl_xor(sd, m, 32);
    }
    if (lane == 0) { s_src[row] = ss; s_dst[row] = sd; }
}

// ---------------------------------------------------------------------------
// Kernel: colsum init to EPS
// ---------------------------------------------------------------------------
__global__ void init_colsum(float* __restrict__ colsum) {
    int t = blockIdx.x * 256 + threadIdx.x;
    colsum[t] = EPS_;
}

// ---------------------------------------------------------------------------
// Kernel: P[b,i,j] = bf16( exp(leaky(s_src[i]+s_dst[j]+bias)) * A[b,i,j] )
//         colsum[b,j] += sum_i P   (softmax over dim=1 -> per-COLUMN sums)
// ---------------------------------------------------------------------------
__global__ void exp_colsum(const float* __restrict__ A, const float* __restrict__ s_src,
                           const float* __restrict__ s_dst, const float* __restrict__ a_b,
                           __bf16* __restrict__ P, float* __restrict__ colsum) {
    __shared__ float ssh[512];
    __shared__ float red[256];
    int b  = blockIdx.z;
    int jt = blockIdx.x;
    int i0 = blockIdx.y * 512;

    for (int t = threadIdx.x; t < 512; t += 256)
        ssh[t] = s_src[b * N_ + i0 + t];
    __syncthreads();

    int j    = jt * 64 + (threadIdx.x & 63);
    int ioff = threadIdx.x >> 6;
    float sd = s_dst[b * N_ + j] + a_b[0];

    const float* Ab = A + (size_t)b * N_ * N_;
    __bf16* Pb = P + (size_t)b * N_ * N_;

    float acc = 0.f;
    for (int ii = ioff; ii < 512; ii += 4) {
        int i = i0 + ii;
        float e = ssh[ii] + sd;
        e = e > 0.f ? e : NEG_SLOPE * e;
        float p = __expf(e) * Ab[(size_t)i * N_ + j];
        Pb[(size_t)i * N_ + j] = (__bf16)p;
        acc += p;
    }
    red[threadIdx.x] = acc;
    __syncthreads();
    if (threadIdx.x < 64) {
        float s = red[threadIdx.x] + red[threadIdx.x + 64] +
                  red[threadIdx.x + 128] + red[threadIdx.x + 192];
        atomicAdd(&colsum[b * N_ + j], s);
    }
}

// ---------------------------------------------------------------------------
// Kernel: WhS = bf16(Wh / colsum[j]) in WMMA B-fragment layout
// Shape: [B][KT=N/32][OT=FOUT/16][lane 32][elem 16]
// ---------------------------------------------------------------------------
__global__ void scale_wh(const float* __restrict__ Wh, const float* __restrict__ colsum,
                         __bf16* __restrict__ WhS) {
    size_t tid = (size_t)blockIdx.x * 256 + threadIdx.x;
    int e    = tid & 15;
    int lane = (tid >> 4) & 31;
    int ot   = (tid >> 9) & 15;
    int kt   = (tid >> 13) & 63;
    int b    = tid >> 19;
    int o = ot * 16 + (lane & 15);
    int j = kt * 32 + (lane >> 4) * 8 + (e & 7) + (e >> 3) * 16;
    float v = Wh[((size_t)b * N_ + j) * FOUT + o] / colsum[b * N_ + j];
    WhS[tid] = (__bf16)v;
}

// ---------------------------------------------------------------------------
extern "C" void kernel_launch(void* const* d_in, const int* in_sizes, int n_in,
                              void* d_out, int out_size, void* d_ws, size_t ws_size,
                              hipStream_t stream) {
    (void)in_sizes; (void)n_in; (void)out_size; (void)ws_size;
    const float* A   = (const float*)d_in[0];
    const float* x   = (const float*)d_in[1];
    const float* W   = (const float*)d_in[2];
    const float* a_w = (const float*)d_in[3];
    const float* a_b = (const float*)d_in[4];
    float* out = (float*)d_out;

    char* w = (char*)d_ws;
    size_t off = 0;
    float* Wh     = (float*)(w + off); off += (size_t)B_ * N_ * FOUT * 4;   // 16 MB
    float* s_src  = (float*)(w + off); off += (size_t)B_ * N_ * 4;
    float* s_dst  = (float*)(w + off); off += (size_t)B_ * N_ * 4;
    float* colsum = (float*)(w + off); off += (size_t)B_ * N_ * 4;
    __bf16* Wf    = (__bf16*)(w + off); off += (size_t)FIN * FOUT * 2;      // 128 KB
    __bf16* WhS   = (__bf16*)(w + off); off += (size_t)B_ * N_ * FOUT * 2;  // 8 MB
    __bf16* P     = (__bf16*)(w + off); off += (size_t)B_ * N_ * N_ * 2;    // 64 MB

    pack_w<<<(FIN * FOUT) / 256, 256, 0, stream>>>(W, Wf);
    wh_gemm<<<512, 256, 0, stream>>>(x, Wf, Wh);
    row_scores<<<(B_ * N_) / 8, 256, 0, stream>>>(Wh, a_w, s_src, s_dst);
    init_colsum<<<(B_ * N_) / 256, 256, 0, stream>>>(colsum);
    exp_colsum<<<dim3(N_ / 64, 4, B_), 256, 0, stream>>>(A, s_src, s_dst, a_b, P, colsum);
    scale_wh<<<(B_ * N_ * FOUT) / 256, 256, 0, stream>>>(Wh, colsum, WhS);
    // 1024 blocks x 128 threads, 2 KB dynamic LDS (double-buffered A tiles)
    attn_gemm<<<dim3(N_ / 16, B_), 128, 2048, stream>>>(P, WhS, out);
}
